// SharedWeightsDense_59184649339047
// MI455X (gfx1250) — matrix operand order repaired and checked by reading, not compiled, not run on hardware
//
#include <hip/hip_runtime.h>
#include <stddef.h>

// Problem constants (match reference)
#define N_TOK 4096
#define DIM   1024
#define ROWS  10000
#define VOCAB 32000
#define KNZ   2

typedef __attribute__((ext_vector_type(16))) __bf16          v16bf;
typedef __attribute__((ext_vector_type(8)))  float           v8f;
typedef __attribute__((ext_vector_type(4)))  float           v4f;
typedef __attribute__((ext_vector_type(8)))  unsigned short  v8us;
typedef __attribute__((ext_vector_type(4)))  int             v4i;

union FragBF {
    v16bf bf;      // 32 bytes = 8 VGPRs, WMMA A/B operand
    v8us  us[2];   // two 16-byte halves
};

// CDNA5 async global->LDS DMA (ASYNCcnt) if the toolchain exposes it.
#if defined(__has_builtin)
#  if __has_builtin(__builtin_amdgcn_global_load_async_to_lds_b128) && \
      __has_builtin(__builtin_amdgcn_s_wait_asynccnt)
#    define USE_ASYNC_LDS 1
#  endif
#endif
#ifndef USE_ASYNC_LDS
#  define USE_ASYNC_LDS 0
#endif

#if USE_ASYNC_LDS
__device__ __forceinline__ void async_copy_b128(const unsigned short* g, unsigned short* l) {
    // Builtin signature (from compiler diagnostic): param0 = v4i in addrspace(1),
    // param1 = v4i in addrspace(3), then imm offset and imm cpol.
    __attribute__((address_space(1))) v4i* gp = (__attribute__((address_space(1))) v4i*)g;
    __attribute__((address_space(3))) v4i* lp = (__attribute__((address_space(3))) v4i*)l;
    __builtin_amdgcn_global_load_async_to_lds_b128(gp, lp, 0, 0);
}
#endif

__device__ __forceinline__ unsigned short f32_to_bf16(float f) {
    union { float f; unsigned int u; } c; c.f = f;
    unsigned int u = c.u;
    u += 0x7FFFu + ((u >> 16) & 1u);    // round-to-nearest-even
    return (unsigned short)(u >> 16);
}

// ---------------------------------------------------------------------------
// Prep 1: W_eff[v,d] = val0 * W[i0,d] + val1 * W[i1,d], stored as bf16.
// ---------------------------------------------------------------------------
__global__ void build_weff_kernel(const float* __restrict__ W,
                                  const int*   __restrict__ code_idx,
                                  const float* __restrict__ code_val,
                                  unsigned short* __restrict__ weff) {
    const int t  = blockIdx.x * blockDim.x + threadIdx.x;   // 0 .. V*D/8-1
    const int v  = t >> 7;                                  // D/8 = 128 per row
    const int dc = (t & 127) << 3;

    const int   i0 = code_idx[v * KNZ + 0];
    const int   i1 = code_idx[v * KNZ + 1];
    const float s0 = code_val[v * KNZ + 0];
    const float s1 = code_val[v * KNZ + 1];

    const v4f* w0 = (const v4f*)(W + (size_t)i0 * DIM + dc);
    const v4f* w1 = (const v4f*)(W + (size_t)i1 * DIM + dc);
    v4f lo = s0 * w0[0] + s1 * w1[0];
    v4f hi = s0 * w0[1] + s1 * w1[1];

    v8us o;
    o[0] = f32_to_bf16(lo[0]); o[1] = f32_to_bf16(lo[1]);
    o[2] = f32_to_bf16(lo[2]); o[3] = f32_to_bf16(lo[3]);
    o[4] = f32_to_bf16(hi[0]); o[5] = f32_to_bf16(hi[1]);
    o[6] = f32_to_bf16(hi[2]); o[7] = f32_to_bf16(hi[3]);
    *(v8us*)(weff + (size_t)t * 8) = o;
}

// ---------------------------------------------------------------------------
// Prep 2: ins (f32) -> bf16.
// ---------------------------------------------------------------------------
__global__ void convert_a_kernel(const float* __restrict__ ins,
                                 unsigned short* __restrict__ a_bf) {
    const int t = blockIdx.x * blockDim.x + threadIdx.x;
    const v4f* s = (const v4f*)(ins + (size_t)t * 8);
    v4f lo = s[0], hi = s[1];
    v8us o;
    o[0] = f32_to_bf16(lo[0]); o[1] = f32_to_bf16(lo[1]);
    o[2] = f32_to_bf16(lo[2]); o[3] = f32_to_bf16(lo[3]);
    o[4] = f32_to_bf16(hi[0]); o[5] = f32_to_bf16(hi[1]);
    o[6] = f32_to_bf16(hi[2]); o[7] = f32_to_bf16(hi[3]);
    *(v8us*)(a_bf + (size_t)t * 8) = o;
}

// ---------------------------------------------------------------------------
// Prep 3: b_eff[v] = val0*b[i0] + val1*b[i1]  (f32)
// ---------------------------------------------------------------------------
__global__ void build_beff_kernel(const float* __restrict__ b,
                                  const int*   __restrict__ code_idx,
                                  const float* __restrict__ code_val,
                                  float* __restrict__ beff) {
    const int v = blockIdx.x * blockDim.x + threadIdx.x;
    if (v < VOCAB) {
        beff[v] = code_val[v * KNZ + 0] * b[code_idx[v * KNZ + 0]] +
                  code_val[v * KNZ + 1] * b[code_idx[v * KNZ + 1]];
    }
}

// ---------------------------------------------------------------------------
// Main GEMM: out[n,v] = sum_d A[n,d] * Weff[v,d] + beff[v]
//
// Block = 256 threads = 8 wave32s; 128(M) x 128(N) output tile.
//   Wave tile = 32x64 = 2(M) x 4(N) accumulator tiles of 16x16.
// K-step = 64 (two 16x16x32 WMMA sub-steps per staged tile).
// B tile (16 KB/step) staged in LDS once, double-buffered.
//   Async path: GLOBAL_LOAD_ASYNC_TO_LDS_B128 issued at the top of each step
//     for the *next* tile -> DMA overlaps the 16 WMMAs; one barrier per step
//     (s_wait_asynccnt 0 + s_barrier).
//   Fallback:  global->VGPR->ds_store with two barriers per step.
// LDS column stride = 64+8 halfs (144 B -> bank stride 36, conflict-free).
// ---------------------------------------------------------------------------
#define KSTEP    64
#define NSTEPS   (DIM / KSTEP)          // 16
#define BSTRIDE  72                     // halfs per column in LDS (64 + 8 pad)
#define BUFHALFS (128 * BSTRIDE)

__global__ void __launch_bounds__(256)
gemm_bf16_kernel(const unsigned short* __restrict__ a_bf,
                 const unsigned short* __restrict__ weff,
                 const float* __restrict__ beff,
                 float* __restrict__ out) {
    __shared__ alignas(16) unsigned short ldsB[2][BUFHALFS];

    const int tid   = threadIdx.x;
    const int wave  = tid >> 5;
    const int lane  = tid & 31;
    const int half  = lane >> 4;         // 0 or 1
    const int l16   = lane & 15;
    const int waveM = wave >> 1;         // 0..3
    const int waveN = wave & 1;          // 0..1

    const int n0 = blockIdx.x * 128 + waveM * 32;  // token rows for this wave
    const int v0 = blockIdx.y * 128;               // vocab cols for this block

    // Cooperative B staging: 2 threads per column, 32 halfs (64 B) each.
    const int bcol = tid >> 1;                     // 0..127
    const int bco  = (tid & 1) * 32;               // half-offset within the 64
    const unsigned short* bgsrc = weff + (size_t)(v0 + bcol) * DIM + bco;
    unsigned short* blds = &ldsB[0][0] + bcol * BSTRIDE + bco;

    // A fragment bases (ISA wave32 16-bit A 16x32 layout).
    const unsigned short* aptr0 = a_bf + (size_t)(n0 + l16) * DIM + half * 8;
    const unsigned short* aptr1 = a_bf + (size_t)(n0 + 16 + l16) * DIM + half * 8;

    // B fragment base in LDS (ISA wave32 16-bit B 32x16 layout).
    const int bfragBase = (waveN * 64 + l16) * BSTRIDE + half * 8;

    v8f c[2][4];
#pragma unroll
    for (int mi = 0; mi < 2; ++mi)
#pragma unroll
        for (int j = 0; j < 4; ++j) c[mi][j] = (v8f)(0.0f);

    FragBF a_cur[2][2];

    // ---- prologue: stage k-step 0 into buffer 0
#if USE_ASYNC_LDS
#pragma unroll
    for (int i = 0; i < 4; ++i)
        async_copy_b128(bgsrc + i * 8, blds + i * 8);
#pragma unroll
    for (int ks = 0; ks < 2; ++ks) {
        a_cur[0][ks].us[0] = *(const v8us*)(aptr0 + ks * 32);
        a_cur[0][ks].us[1] = *(const v8us*)(aptr0 + ks * 32 + 16);
        a_cur[1][ks].us[0] = *(const v8us*)(aptr1 + ks * 32);
        a_cur[1][ks].us[1] = *(const v8us*)(aptr1 + ks * 32 + 16);
    }
    __builtin_amdgcn_s_wait_asynccnt(0);
    __syncthreads();
#else
    v8us breg[4];
#pragma unroll
    for (int i = 0; i < 4; ++i) breg[i] = *(const v8us*)(bgsrc + i * 8);
#pragma unroll
    for (int ks = 0; ks < 2; ++ks) {
        a_cur[0][ks].us[0] = *(const v8us*)(aptr0 + ks * 32);
        a_cur[0][ks].us[1] = *(const v8us*)(aptr0 + ks * 32 + 16);
        a_cur[1][ks].us[0] = *(const v8us*)(aptr1 + ks * 32);
        a_cur[1][ks].us[1] = *(const v8us*)(aptr1 + ks * 32 + 16);
    }
#pragma unroll
    for (int i = 0; i < 4; ++i) *(v8us*)(blds + i * 8) = breg[i];
    __syncthreads();
#endif

#pragma unroll 2
    for (int step = 0; step < NSTEPS; ++step) {
        const int cur = step & 1;
        const int nxt = cur ^ 1;
        const int k1  = (step + 1) * KSTEP;

#if USE_ASYNC_LDS
        // Issue DMA for the next tile NOW (buffer nxt is idle since the last
        // barrier); it runs concurrently with the 16 WMMAs below.
        if (step + 1 < NSTEPS) {
#pragma unroll
            for (int i = 0; i < 4; ++i)
                async_copy_b128(bgsrc + k1 + i * 8, blds + nxt * BUFHALFS + i * 8);
        }
#endif
        // Prefetch next A fragments (and, in fallback, the next B tile) to regs.
        FragBF a_nxt[2][2];
        if (step + 1 < NSTEPS) {
#if !USE_ASYNC_LDS
#pragma unroll
            for (int i = 0; i < 4; ++i)
                breg[i] = *(const v8us*)(bgsrc + k1 + i * 8);
#endif
#pragma unroll
            for (int ks = 0; ks < 2; ++ks) {
                a_nxt[0][ks].us[0] = *(const v8us*)(aptr0 + k1 + ks * 32);
                a_nxt[0][ks].us[1] = *(const v8us*)(aptr0 + k1 + ks * 32 + 16);
                a_nxt[1][ks].us[0] = *(const v8us*)(aptr1 + k1 + ks * 32);
                a_nxt[1][ks].us[1] = *(const v8us*)(aptr1 + k1 + ks * 32 + 16);
            }
        }

        // Compute: 2 k-substeps x 4 N tiles x 2 M tiles = 16 WMMAs.
        const unsigned short* lb = &ldsB[cur][0] + bfragBase;
#pragma unroll
        for (int ks = 0; ks < 2; ++ks) {
            const int ko = ks * 32;
            FragBF bcur;
            bcur.us[0] = *(const v8us*)(lb + ko);
            bcur.us[1] = *(const v8us*)(lb + ko + 16);
#pragma unroll
            for (int j = 0; j < 4; ++j) {
                FragBF bnxt;
                if (j < 3) {  // double-buffer: load fragment j+1 before using j
                    bnxt.us[0] = *(const v8us*)(lb + (j + 1) * 16 * BSTRIDE + ko);
                    bnxt.us[1] = *(const v8us*)(lb + (j + 1) * 16 * BSTRIDE + ko + 16);
                }
                c[0][j] = __builtin_amdgcn_wmma_f32_16x16x32_bf16(
                    false, a_cur[0][ks].bf, false, bcur.bf, (short)0, c[0][j], false, false);
                c[1][j] = __builtin_amdgcn_wmma_f32_16x16x32_bf16(
                    false, a_cur[1][ks].bf, false, bcur.bf, (short)0, c[1][j], false, false);
                if (j < 3) bcur = bnxt;
            }
        }

        if (step + 1 < NSTEPS) {
#if USE_ASYNC_LDS
            __builtin_amdgcn_s_wait_asynccnt(0);   // own DMA done
            __syncthreads();                       // everyone's DMA done
#else
            __syncthreads();                       // all reads of lds[nxt] done
#pragma unroll
            for (int i = 0; i < 4; ++i)
                *(v8us*)(blds + nxt * BUFHALFS + i * 8) = breg[i];
            __syncthreads();                       // next buffer visible
#endif
#pragma unroll
            for (int ks = 0; ks < 2; ++ks) {
                a_cur[0][ks] = a_nxt[0][ks];
                a_cur[1][ks] = a_nxt[1][ks];
            }
        }
    }

    // Epilogue: C/D 16x16 f32 layout -> VGPR r: lanes 0-15 M=r, lanes 16-31 M=r+8.
#pragma unroll
    for (int mi = 0; mi < 2; ++mi) {
        const int row_base = n0 + mi * 16 + half * 8;
#pragma unroll
        for (int j = 0; j < 4; ++j) {
            const int col = v0 + waveN * 64 + j * 16 + l16;
            const float be = beff[col];
            size_t base = (size_t)row_base * VOCAB + col;
#pragma unroll
            for (int r = 0; r < 8; ++r) {
                out[base] = c[mi][j][r] + be;
                base += VOCAB;
            }
        }
    }
}

// ---------------------------------------------------------------------------
extern "C" void kernel_launch(void* const* d_in, const int* in_sizes, int n_in,
                              void* d_out, int out_size, void* d_ws, size_t ws_size,
                              hipStream_t stream) {
    const float* ins      = (const float*)d_in[0];   // [N, D]
    const float* W        = (const float*)d_in[1];   // [R, D]
    const float* b        = (const float*)d_in[2];   // [R]
    const int*   code_idx = (const int*)  d_in[3];   // [V, K]
    const float* code_val = (const float*)d_in[4];   // [V, K]
    float*       out      = (float*)d_out;           // [N, V]

    // Workspace layout: weff bf16 [V*D] | a_bf bf16 [N*D] | beff f32 [V]
    unsigned short* weff = (unsigned short*)d_ws;
    unsigned short* a_bf = weff + (size_t)VOCAB * DIM;
    float*          beff = (float*)(a_bf + (size_t)N_TOK * DIM);

    {   // W_eff: V*D/8 threads
        const int total = VOCAB * DIM / 8;
        build_weff_kernel<<<total / 256, 256, 0, stream>>>(W, code_idx, code_val, weff);
    }
    {   // A bf16: N*D/8 threads
        const int total = N_TOK * DIM / 8;
        convert_a_kernel<<<total / 256, 256, 0, stream>>>(ins, a_bf);
    }
    build_beff_kernel<<<(VOCAB + 255) / 256, 256, 0, stream>>>(b, code_idx, code_val, beff);

    dim3 grid(N_TOK / 128, VOCAB / 128);   // 32 x 250
    gemm_bf16_kernel<<<grid, 256, 0, stream>>>(a_bf, weff, beff, out);
}
